// VectorQuantizer1D_74242804678713
// MI455X (gfx1250) — matrix operand order repaired and checked by reading, not compiled, not run on hardware
//
#include <hip/hip_runtime.h>
#include <hip/hip_bf16.h>
#include <math.h>

// ---------------------------------------------------------------------------
// VQ-VAE vector quantizer for MI455X (gfx1250, wave32, WMMA).
//   inputs: [B=16, C=64, T=2048] f32 ; weight: [K=1024, D=64] f32
//   outputs (flat f32): loss(1) | quant_out(16*64*2048) | perplexity(1) | encodings(32768*1024)
// Distance GEMM done with split-f16 WMMA (hi/lo) for ~fp32-accurate argmin.
// ---------------------------------------------------------------------------

#define KC   1024            // num codes
#define DD   64              // embedding dim (== C)
#define BB   16
#define TT   2048
#define NR   (BB * TT)       // 32768 rows

typedef _Float16 f16;
typedef __attribute__((ext_vector_type(16))) _Float16 v16h;
typedef __attribute__((ext_vector_type(8)))  float    v8f;

// ---------------------------------------------------------------------------
// Kernel 0: split weights into f16 hi/lo, compute 0.5*||e_j||^2, zero counts.
// grid = KC blocks of 64 threads (one code row per block).
// ---------------------------------------------------------------------------
__global__ __launch_bounds__(64)
void vq_prep(const float* __restrict__ w, f16* __restrict__ w_hi,
             f16* __restrict__ w_lo, float* __restrict__ e2,
             int* __restrict__ counts) {
  const int j = blockIdx.x;
  const int d = threadIdx.x;
  float v  = w[j * DD + d];
  f16  hi  = (f16)v;
  f16  lo  = (f16)(v - (float)hi);
  w_hi[j * DD + d] = hi;
  w_lo[j * DD + d] = lo;

  __shared__ float red[64];
  red[d] = v * v;
  __syncthreads();
  if (d < 32) {
    float s = red[d] + red[d + 32];
    for (int off = 16; off > 0; off >>= 1) s += __shfl_down(s, off);
    if (d == 0) e2[j] = 0.5f * s;
  }
  if (d == 0) counts[j] = 0;
}

// ---------------------------------------------------------------------------
// Kernel 1: per-row argmin over K codes via split-f16 WMMA.
// score_j = 0.5*||e_j||^2 + (-x)·e_j   (monotone in true distance per row)
// 256 threads = 8 waves; each wave owns 16 rows; WG covers 128 rows.
// ---------------------------------------------------------------------------
__global__ __launch_bounds__(256)
void vq_argmin(const float* __restrict__ x, const f16* __restrict__ w_hi,
               const f16* __restrict__ w_lo, const float* __restrict__ e2,
               int* __restrict__ idx_out, int* __restrict__ counts) {
  const int tid  = threadIdx.x;
  const int lane = tid & 31;
  const int wave = tid >> 5;
  const int half = lane >> 4;     // 0: lanes 0-15, 1: lanes 16-31
  const int col  = lane & 15;
  const int rowBase = blockIdx.x * 128 + wave * 16;

  // --- Build A fragments: 16x64 tile of (-x) split into f16 hi/lo.
  // 16-bit A layout used here, same k order as B (16x32): lane L holds row M=L%16;
//   elems 0..15 -> K = i + 16*half
  v16h a_hi[2], a_lo[2];
  {
    const int n = rowBase + col;      // this lane's row (M = lane % 16)
    const int b = n >> 11;
    const int t = n & 2047;
    const float* xrow = x + (size_t)b * (DD * TT) + t;   // + d*TT per dim
#pragma unroll
    for (int c = 0; c < 2; ++c) {
#pragma unroll
      for (int i = 0; i < 16; ++i) {
        const int kl = i + 16 * half;
        const int d  = 32 * c + kl;
        float v = -xrow[(size_t)d * TT];     // negate: fold -x·e into WMMA add
        f16 hi = (f16)v;
        f16 lo = (f16)(v - (float)hi);
        a_hi[c][i] = hi;
        a_lo[c][i] = lo;
      }
    }
  }

  // Running argmin per (row r in C-layout, col-class = lane%16).
  float best[8];
  int   bestj[8];
#pragma unroll
  for (int r = 0; r < 8; ++r) { best[r] = 3.0e38f; bestj[r] = 0; }

  for (int jt = 0; jt < KC / 16; ++jt) {
    const int   j    = jt * 16 + col;
    const float bias = e2[j];                 // 0.5*||e_j||^2, per column N
    v8f acc;
#pragma unroll
    for (int r = 0; r < 8; ++r) acc[r] = bias;

#pragma unroll
    for (int c = 0; c < 2; ++c) {
      // B layout (32x16, K x N): lane holds column N=lane%16,
      // elems 0..15 = K = 16*half + i  -> 16 contiguous f16 = one 32B load.
      const size_t off = (size_t)j * DD + 32 * c + 16 * half;
      const v16h bh = *(const v16h*)(w_hi + off);
      const v16h bl = *(const v16h*)(w_lo + off);
      acc = __builtin_amdgcn_wmma_f32_16x16x32_f16(false, a_hi[c], false, bh,
                                                   (short)0, acc, false, false);
      acc = __builtin_amdgcn_wmma_f32_16x16x32_f16(false, a_hi[c], false, bl,
                                                   (short)0, acc, false, false);
      acc = __builtin_amdgcn_wmma_f32_16x16x32_f16(false, a_lo[c], false, bh,
                                                   (short)0, acc, false, false);
    }
    // Branchless running min (strict <: keeps first/lowest index in-class).
#pragma unroll
    for (int r = 0; r < 8; ++r) {
      const bool lt = acc[r] < best[r];
      best[r]  = lt ? acc[r] : best[r];
      bestj[r] = lt ? j      : bestj[r];
    }
  }

  // Cross-lane reduce over 16 column classes. C layout: VGPR r, lane L:
  // row M = r + 8*(L/16), col-class = L%16.
  __shared__ float sBest[8][16][17];
  __shared__ int   sIdx[8][16][17];
#pragma unroll
  for (int r = 0; r < 8; ++r) {
    const int row = r + 8 * half;
    sBest[wave][row][col] = best[r];
    sIdx[wave][row][col]  = bestj[r];
  }
  __syncthreads();
  if (lane < 16) {
    float bv = sBest[wave][lane][0];
    int   bj = sIdx[wave][lane][0];
#pragma unroll
    for (int c2 = 1; c2 < 16; ++c2) {
      const float v  = sBest[wave][lane][c2];
      const int   j2 = sIdx[wave][lane][c2];
      if (v < bv || (v == bv && j2 < bj)) { bv = v; bj = j2; }  // first-index ties
    }
    idx_out[rowBase + lane] = bj;
    atomicAdd(&counts[bj], 1);     // integer atomic: deterministic
  }
}

// ---------------------------------------------------------------------------
// Kernel 2: one-hot encodings (b128 stores), straight-through quant_out,
// per-row SSE partial. One block per row n.
// ---------------------------------------------------------------------------
__global__ __launch_bounds__(256)
void vq_out(const float* __restrict__ x, const float* __restrict__ w,
            const int* __restrict__ idx, float* __restrict__ quant,
            float* __restrict__ enc, float* __restrict__ partials) {
  const int n   = blockIdx.x;
  const int tid = threadIdx.x;
  const int jn  = idx[n];

  // encodings row: 1024 floats, 4 per thread, mostly zeros + one 1.0.
  float4 v4 = make_float4(0.f, 0.f, 0.f, 0.f);
  const int base = tid * 4;
  if (jn >= base && jn < base + 4) ((float*)&v4)[jn - base] = 1.0f;
  *(float4*)(enc + (size_t)n * KC + base) = v4;

  __shared__ float red[64];
  if (tid < DD) {
    const int b = n >> 11, t = n & 2047;
    const size_t xoff = (size_t)b * (DD * TT) + (size_t)tid * TT + t;
    const float q  = w[jn * DD + tid];   // forward value of straight-through
    const float xv = x[xoff];
    quant[xoff] = q;
    const float diff = q - xv;
    red[tid] = diff * diff;
  }
  __syncthreads();
  if (tid < 32) {
    float s = red[tid] + red[tid + 32];
    for (int off = 16; off > 0; off >>= 1) s += __shfl_down(s, off);
    if (tid == 0) partials[n] = s;       // fixed location: deterministic
  }
}

// ---------------------------------------------------------------------------
// Kernel 3: deterministic final reductions -> loss, perplexity.
// ---------------------------------------------------------------------------
__global__ __launch_bounds__(256)
void vq_finalize(const float* __restrict__ partials, const int* __restrict__ counts,
                 float* __restrict__ out_loss, float* __restrict__ out_perp) {
  const int tid = threadIdx.x;
  __shared__ float red[256];

  float s = 0.f;
  for (int i = tid; i < NR; i += 256) s += partials[i];
  red[tid] = s;
  __syncthreads();
  for (int st = 128; st > 0; st >>= 1) {
    if (tid < st) red[tid] += red[tid + st];
    __syncthreads();
  }
  if (tid == 0) *out_loss = 1.25f * red[0] / (float)(NR * DD);
  __syncthreads();

  float e = 0.f;
  for (int k = tid; k < KC; k += 256) {
    const float p = (float)counts[k] / (float)NR;
    e -= p * logf(p + 1e-10f);
  }
  red[tid] = e;
  __syncthreads();
  for (int st = 128; st > 0; st >>= 1) {
    if (tid < st) red[tid] += red[tid + st];
    __syncthreads();
  }
  if (tid == 0) *out_perp = expf(red[0]);
}

// ---------------------------------------------------------------------------
extern "C" void kernel_launch(void* const* d_in, const int* in_sizes, int n_in,
                              void* d_out, int out_size, void* d_ws, size_t ws_size,
                              hipStream_t stream) {
  (void)in_sizes; (void)n_in; (void)out_size; (void)ws_size;
  const float* x = (const float*)d_in[0];   // inputs [16,64,2048]
  const float* w = (const float*)d_in[1];   // weight [1024,64]

  float* out      = (float*)d_out;
  float* out_loss = out;                          // [0]
  float* quant    = out + 1;                      // [1 .. 2097152]
  float* out_perp = out + 1 + (size_t)BB * DD * TT;       // [2097153]
  float* enc      = out + 2 + (size_t)BB * DD * TT;       // encodings [N,K]

  char* ws = (char*)d_ws;
  int*   idx      = (int*)(ws + 0);          // 32768 ints   (128 KB)
  int*   counts   = (int*)(ws + 131072);     // 1024 ints    (4 KB)
  float* e2       = (float*)(ws + 135168);   // 1024 f32     (4 KB)
  f16*   w_hi     = (f16*)(ws + 139264);     // 65536 f16    (128 KB, 32B aligned)
  f16*   w_lo     = (f16*)(ws + 270336);     // 65536 f16    (128 KB, 32B aligned)
  float* partials = (float*)(ws + 401408);   // 32768 f32    (128 KB)

  vq_prep<<<KC, 64, 0, stream>>>(w, w_hi, w_lo, e2, counts);
  vq_argmin<<<NR / 128, 256, 0, stream>>>(x, w_hi, w_lo, e2, idx, counts);
  vq_out<<<NR, 256, 0, stream>>>(x, w, idx, quant, enc, partials);
  vq_finalize<<<1, 256, 0, stream>>>(partials, counts, out_loss, out_perp);
}
